// Get_style_components_32719060861568
// MI455X (gfx1250) — matrix operand order repaired and checked by reading, not compiled, not run on hardware
//
#include <hip/hip_runtime.h>

// ---------------------------------------------------------------------------
// Fused multi-head cross-attention + residual + LayerNorm for MI455X (gfx1250)
// B=64, N=100, C=256, heads=8, d=32, L=3*16*16=768.
// Roofline: 215MB mandatory HBM traffic @ 23.3TB/s ~ 9.2us; 19.6 GFLOP needs
// >2 PFLOPS effective -> f16 WMMA (V_WMMA_F32_16X16X32_F16), f32 accumulate.
// Scores (39M floats) never touch HBM: fused flash-style online softmax.
// K-slice staged to LDS via the Tensor Data Mover (tensor_load_to_lds).
// ---------------------------------------------------------------------------

typedef _Float16 h16;
typedef __attribute__((ext_vector_type(16))) _Float16 v16h;
typedef __attribute__((ext_vector_type(8)))  _Float16 v8h;
typedef __attribute__((ext_vector_type(8)))  float    v8f;
typedef __attribute__((ext_vector_type(4)))  float    v4f;
typedef __attribute__((ext_vector_type(4)))  unsigned int u32x4;
typedef __attribute__((ext_vector_type(8)))  int      i32x8;
typedef __attribute__((ext_vector_type(4)))  int      i32x4;

#define BATCH 64
#define SEQ_N 100
#define CH    256
#define HEADS 8
#define HDIM  32
#define LREF  768                 // 3 * 16 * 16
#define BN    (BATCH * SEQ_N)     // 6400
#define BL    (BATCH * LREF)      // 49152

#if defined(__has_builtin)
#if __has_builtin(__builtin_amdgcn_tensor_load_to_lds) && \
    __has_builtin(__builtin_amdgcn_s_wait_tensorcnt)
#define USE_TDM 1
#endif
#endif

__device__ __forceinline__ v8f wmma_f16(v16h a, v16h b, v8f c) {
  // D = A(16x32 f16) x B(32x16 f16) + C(16x16 f32)
  return __builtin_amdgcn_wmma_f32_16x16x32_f16(
      /*neg_a=*/false, a, /*neg_b=*/false, b,
      /*c_mod=*/(short)0, c, /*reuse_a=*/false, /*reuse_b=*/false);
}

// A operand (16x32, f16): lane m=lane&15 holds row m; lane-half hi selects
// K sub-blocks [hi*8, hi*8+8) and [16+hi*8, 16+hi*8+8)  (ISA 7.12.2).
__device__ __forceinline__ v16h load_a(const h16* base, int ld) {
  const int lane = threadIdx.x & 31;
  const int m = lane & 15, hi = lane >> 4;
  const h16* p = base + m * ld + hi * 8;
  v8h c0 = *(const v8h*)(p);
  v8h c1 = *(const v8h*)(p + 16);
  v16h r;
#pragma unroll
  for (int i = 0; i < 8; ++i) { r[i] = c0[i]; r[i + 8] = c1[i]; }
  return r;
}

// B operand (32x16, f16): lane n=lane&15 is the output column; lane-half hi
// selects K sub-blocks [hi*8, hi*8+8) and [16+hi*8, 16+hi*8+8).
__device__ __forceinline__ v16h load_b(const h16* base, int ld) {
  const int lane = threadIdx.x & 31;
  const int n = lane & 15, hi = lane >> 4;
  const h16* p = base + n * ld + hi * 8;
  v8h c0 = *(const v8h*)(p);
  v8h c1 = *(const v8h*)(p + 16);
  v16h r;
#pragma unroll
  for (int i = 0; i < 8; ++i) { r[i] = c0[i]; r[i + 8] = c1[i]; }
  return r;
}

// A operand from f32 row-major memory (converts on the fly).
__device__ __forceinline__ v16h load_a_f32(const float* base, int ld) {
  const int lane = threadIdx.x & 31;
  const int m = lane & 15, hi = lane >> 4;
  const float* p = base + m * ld + hi * 8;
  v16h r;
#pragma unroll
  for (int i = 0; i < 8; ++i) { r[i] = (h16)p[i]; r[i + 8] = (h16)p[16 + i]; }
  return r;
}

// ---------------------------------------------------------------------------
// Kernel 1: convert weight matrices to f16 once (L2-resident afterwards).
// ---------------------------------------------------------------------------
__global__ void wcvt_kernel(const float* Wq, const float* Wk, const float* Wv,
                            const float* Wo, h16* hq, h16* hk, h16* hv, h16* ho) {
  const int i = blockIdx.x * blockDim.x + threadIdx.x;
  if (i < CH * CH) {
    hq[i] = (h16)Wq[i]; hk[i] = (h16)Wk[i];
    hv[i] = (h16)Wv[i]; ho[i] = (h16)Wo[i];
  }
}

// ---------------------------------------------------------------------------
// Kernel 2: Q = comp_seq @ Wq^T  -> f16 [BN][CH].
// Block = 4 waves; each wave: 16 rows x 64 cols, K-loop over 256 in steps of 32.
// ---------------------------------------------------------------------------
__global__ void qproj_kernel(const float* __restrict__ X,
                             const h16* __restrict__ W, h16* __restrict__ Q) {
  const int wave = threadIdx.x >> 5;
  const int lane = threadIdx.x & 31;
  const int m0 = blockIdx.x * 16;     // 6400/16 = 400 exact
  const int n0 = wave * 64;
  v8f acc[4] = {};
#pragma unroll
  for (int kk = 0; kk < CH; kk += 32) {
    v16h a = load_a_f32(X + (size_t)m0 * CH + kk, CH);
#pragma unroll
    for (int j = 0; j < 4; ++j) {
      v16h b = load_b(W + (size_t)(n0 + j * 16) * CH + kk, CH);  // B[k][n] = W[n][k]
      acc[j] = wmma_f16(a, b, acc[j]);
    }
  }
  const int hi = lane >> 4, c0 = n0 + (lane & 15);
#pragma unroll
  for (int j = 0; j < 4; ++j)
#pragma unroll
    for (int r = 0; r < 8; ++r)
      Q[(size_t)(m0 + r + 8 * hi) * CH + c0 + j * 16] = (h16)acc[j][r];
}

// ---------------------------------------------------------------------------
// Kernel 3: K/V projections over the gathered ref sequence.
// 64-row blocks, 8 waves: waves 0-3 compute K, waves 4-7 compute V; each wave
// owns a 64x64 tile (16 WMMA accumulators) so every B-operand fetch feeds 4
// WMMAs -> weight L2 traffic drops to ~196MB total.
// ref_seq[b,l,c] = ref[b,kpl,c,hw] (channel-major) -> transposed through LDS.
// Outputs: Kh f16 [BL][CH];  Vt f16 [B][heads][d][LREF] (transposed so the
// attention PV B-operand is two 16B reads per lane).
// ---------------------------------------------------------------------------
__global__ void kvproj_kernel(const float* __restrict__ R,
                              const h16* __restrict__ Wk,
                              const h16* __restrict__ Wv,
                              h16* __restrict__ Kh, h16* __restrict__ Vt) {
  __shared__ h16 Xs[64 * 40];                 // 64 rows x 32 ch, stride 40 (16B-aligned)
  const int wave = threadIdx.x >> 5;
  const int lane = threadIdx.x & 31;
  const int l0g  = blockIdx.x * 64;           // 49152/64 = 768 blocks, same ref plane
  const int b    = l0g / LREF;
  const int lrel = l0g - b * LREF;
  const int kpl  = lrel >> 8;                 // which of the 3 reference maps
  const int hw0  = lrel & 255;                // 64-aligned within the 256-pixel plane
  const bool isV = wave >= 4;
  const int n0   = (wave & 3) * 64;
  const h16* W   = isV ? Wv : Wk;
  const float* src = R + (size_t)(b * 3 + kpl) * CH * 256;   // + c*256 + hw
  v8f acc[4][4] = {};                          // [row strip][col tile]
  const int cs = threadIdx.x >> 3;             // staged channel 0..31
  const int ps = threadIdx.x & 7;              // 8-row part 0..7

  for (int kk = 0; kk < CH; kk += 32) {
    __syncthreads();
    const float* sp = src + (size_t)(kk + cs) * 256 + hw0 + ps * 8;
    v4f d0 = *(const v4f*)(sp);
    v4f d1 = *(const v4f*)(sp + 4);
#pragma unroll
    for (int i = 0; i < 4; ++i) {
      Xs[(ps * 8 + i) * 40 + cs]     = (h16)d0[i];   // transpose hw-major -> row-major
      Xs[(ps * 8 + 4 + i) * 40 + cs] = (h16)d1[i];
    }
    __syncthreads();
    v16h a[4];
#pragma unroll
    for (int s2 = 0; s2 < 4; ++s2) a[s2] = load_a(Xs + s2 * 16 * 40, 40);
#pragma unroll
    for (int j = 0; j < 4; ++j) {
      v16h bw = load_b(W + (size_t)(n0 + j * 16) * CH + kk, CH);
#pragma unroll
      for (int s2 = 0; s2 < 4; ++s2) acc[s2][j] = wmma_f16(a[s2], bw, acc[s2][j]);
    }
  }
  const int hi = lane >> 4;
  if (!isV) {
#pragma unroll
    for (int s2 = 0; s2 < 4; ++s2)
#pragma unroll
      for (int j = 0; j < 4; ++j) {
        const int col = n0 + j * 16 + (lane & 15);
#pragma unroll
        for (int r = 0; r < 8; ++r)
          Kh[(size_t)(l0g + s2 * 16 + r + 8 * hi) * CH + col] = (h16)acc[s2][j][r];
      }
  } else {
#pragma unroll
    for (int s2 = 0; s2 < 4; ++s2)
#pragma unroll
      for (int j = 0; j < 4; ++j) {
        const int col = n0 + j * 16 + (lane & 15);
        v8h pv;                                 // rows r are consecutive l -> 16B store
#pragma unroll
        for (int r = 0; r < 8; ++r) pv[r] = (h16)acc[s2][j][r];
        h16* dst = Vt + (size_t)((b * HEADS + (col >> 5)) * HDIM + (col & 31)) * LREF
                      + lrel + s2 * 16 + 8 * hi;
        *(v8h*)dst = pv;
      }
  }
}

// ---------------------------------------------------------------------------
// Kernel 4: fused attention. Grid = B*heads*2: each block owns (b, head) and
// up to 4 of the 7 query strips. S = Q K^T (no 1/sqrt(d) scale per reference),
// online softmax over L=768, out = P V. K slice (48KB) staged into LDS by the
// Tensor Data Mover (2D D#: 32 x 768 tile of 2B elements, row stride 256);
// V^T streamed from L2 (25MB tensor is L2-resident). P tiles convert
// C-layout -> A-operand layout via 1KB/wave LDS (same-wave DS ops in-order).
// ---------------------------------------------------------------------------
__global__ void attn_kernel(const h16* __restrict__ Qh, const h16* __restrict__ Kh,
                            const h16* __restrict__ Vt, h16* __restrict__ O) {
  __shared__ h16 sm[LREF * HDIM + 4 * 512];   // 48KB K + 4x1KB P staging
  h16* Ks = sm;
  const int wave = threadIdx.x >> 5;
  const int lane = threadIdx.x & 31;
  h16* Ps = sm + LREF * HDIM + wave * 512;
  const int b    = blockIdx.x >> 4;
  const int rem  = blockIdx.x & 15;
  const int hd   = rem >> 1;
  const int half = rem & 1;                   // strips [0..3] or [4..6]
  const h16* ksrc = Kh + (size_t)b * LREF * CH + hd * HDIM;

#ifdef USE_TDM
  if (wave == 0) {
    const unsigned long long ga = (unsigned long long)(const void*)ksrc;
    const unsigned lds_off = (unsigned)(unsigned long long)(void*)Ks;
    u32x4 g0;
    g0[0] = 1u;                                        // count=1 (valid user D#)
    g0[1] = lds_off;                                   // lds_addr (bytes)
    g0[2] = (unsigned)(ga & 0xffffffffu);              // global_addr[31:0]
    g0[3] = (unsigned)((ga >> 32) & 0x1ffffffu)        // global_addr[56:32]
            | 0x80000000u;                             // type = 2 ("image")
    i32x8 g1;
    g1[0] = 0x00010000;                                // data_size = 2B
    g1[1] = (int)(32u << 16);                          // tensor_dim0 = 32
    g1[2] = (int)(768u << 16);                         // tensor_dim1 = 768
    g1[3] = (int)(32u << 16);                          // tile_dim0 = 32
    g1[4] = 768;                                       // tile_dim1=768, tile_dim2=0
    g1[5] = 256;                                       // tensor_dim0_stride = 256
    g1[6] = 0;                                         // dim1_stride (unused, 2D)
    g1[7] = 0;
    const i32x4 z4 = {0, 0, 0, 0};
#if __clang_major__ >= 23
    const i32x8 z8 = {0, 0, 0, 0, 0, 0, 0, 0};
    __builtin_amdgcn_tensor_load_to_lds(g0, g1, z4, z4, z8, 0);
#else
    __builtin_amdgcn_tensor_load_to_lds(g0, g1, z4, z4, 0);
#endif
    __builtin_amdgcn_s_wait_tensorcnt(0);
  }
#else
  for (int idx = threadIdx.x; idx < LREF * 2; idx += blockDim.x) {
    const int l = idx >> 1, hc = (idx & 1) * 16;
    *(v16h*)(Ks + l * HDIM + hc) = *(const v16h*)(ksrc + (size_t)l * CH + hc);
  }
#endif
  __syncthreads();

  const h16* vsrc = Vt + (size_t)(b * HEADS + hd) * HDIM * LREF;  // [32][768]
  const int m = lane & 15, hi = lane >> 4;
  const int s = half * 4 + wave;                // this wave's query strip

  if (s < 7) {
    int qrow = b * SEQ_N + s * 16 + m;
    if (qrow > BN - 1) qrow = BN - 1;           // clamp pad rows (rowwise -> harmless)
    const h16* qp = Qh + (size_t)qrow * CH + hd * HDIM + hi * 8;
    v8h q0 = *(const v8h*)(qp);
    v8h q1 = *(const v8h*)(qp + 16);
    v16h aq;
#pragma unroll
    for (int i = 0; i < 8; ++i) { aq[i] = q0[i]; aq[i + 8] = q1[i]; }

    v8f acc0 = {}, acc1 = {};                   // out tile, d=0..15 / 16..31
    float mrow[8], lsum[8];
#pragma unroll
    for (int r = 0; r < 8; ++r) { mrow[r] = -3.0e38f; lsum[r] = 0.f; }

    for (int l0 = 0; l0 < LREF; l0 += 32) {
      v8f z0 = {}, z1 = {};
      v8f s0 = wmma_f16(aq, load_b(Ks + l0 * HDIM, HDIM), z0);
      v8f s1 = wmma_f16(aq, load_b(Ks + (l0 + 16) * HDIM, HDIM), z1);
#pragma unroll
      for (int r = 0; r < 8; ++r) {             // online softmax, row = r + 8*hi
        float v = fmaxf(s0[r], s1[r]);
#pragma unroll
        for (int off = 1; off < 16; off <<= 1) v = fmaxf(v, __shfl_xor(v, off, 32));
        const float mn = fmaxf(mrow[r], v);
        const float scale = __expf(mrow[r] - mn);
        mrow[r] = mn;
        const float p0 = __expf(s0[r] - mn);
        const float p1 = __expf(s1[r] - mn);
        float rs = p0 + p1;
#pragma unroll
        for (int off = 1; off < 16; off <<= 1) rs += __shfl_xor(rs, off, 32);
        lsum[r] = lsum[r] * scale + rs;
        acc0[r] *= scale; acc1[r] *= scale;
        Ps[(r + 8 * hi) * 32 + m]      = (h16)p0;   // C-layout -> A-layout staging
        Ps[(r + 8 * hi) * 32 + 16 + m] = (h16)p1;
      }
      v16h ap = load_a(Ps, 32);                 // P (16n x 32l)
      acc0 = wmma_f16(ap, load_b(vsrc + l0, LREF), acc0);
      acc1 = wmma_f16(ap, load_b(vsrc + (size_t)16 * LREF + l0, LREF), acc1);
    }
#pragma unroll
    for (int r = 0; r < 8; ++r) {
      const int n = s * 16 + r + 8 * hi;
      if (n < SEQ_N) {
        const float inv = 1.f / lsum[r];
        h16* op = O + (size_t)(b * SEQ_N + n) * CH + hd * HDIM + m;
        op[0]  = (h16)(acc0[r] * inv);
        op[16] = (h16)(acc1[r] * inv);
      }
    }
  }
}

// ---------------------------------------------------------------------------
// Kernel 5: Y = attn_out @ Wo^T + residual, then LayerNorm over C=256.
// Each wave owns 16 complete rows (16 output tiles) so LN is wave-local.
// ---------------------------------------------------------------------------
__global__ void oproj_ln_kernel(const h16* __restrict__ Xa, const h16* __restrict__ Wo,
                                const float* __restrict__ resid,
                                const float* __restrict__ gamma,
                                const float* __restrict__ beta,
                                float* __restrict__ out) {
  const int wave = threadIdx.x >> 5;
  const int lane = threadIdx.x & 31;
  const int m0 = (blockIdx.x * 4 + wave) * 16;   // 400 strips total
  v8f acc[16] = {};
  for (int kk = 0; kk < CH; kk += 32) {
    v16h a = load_a(Xa + (size_t)m0 * CH + kk, CH);
#pragma unroll
    for (int j = 0; j < 16; ++j) {
      v16h b = load_b(Wo + (size_t)(j * 16) * CH + kk, CH);
      acc[j] = wmma_f16(a, b, acc[j]);
    }
  }
  const int m = lane & 15, hi = lane >> 4;
  float sum[8], sq[8];
#pragma unroll
  for (int r = 0; r < 8; ++r) { sum[r] = 0.f; sq[r] = 0.f; }
#pragma unroll
  for (int j = 0; j < 16; ++j) {
    const int col = j * 16 + m;
#pragma unroll
    for (int r = 0; r < 8; ++r) {
      float x = acc[j][r] + resid[(size_t)(m0 + r + 8 * hi) * CH + col];
      acc[j][r] = x;
      sum[r] += x; sq[r] += x * x;
    }
  }
#pragma unroll
  for (int r = 0; r < 8; ++r)
#pragma unroll
    for (int off = 1; off < 16; off <<= 1) {
      sum[r] += __shfl_xor(sum[r], off, 32);
      sq[r]  += __shfl_xor(sq[r],  off, 32);
    }
#pragma unroll
  for (int j = 0; j < 16; ++j) {
    const int col = j * 16 + m;
    const float g = gamma[col], be = beta[col];
#pragma unroll
    for (int r = 0; r < 8; ++r) {
      const float mean = sum[r] * (1.f / CH);
      const float var  = sq[r] * (1.f / CH) - mean * mean;
      out[(size_t)(m0 + r + 8 * hi) * CH + col] =
          (acc[j][r] - mean) * rsqrtf(var + 1e-6f) * g + be;
    }
  }
}

// ---------------------------------------------------------------------------
// Host launcher. Workspace layout (bytes):
//   Whq 0  Whk 131072  Whv 262144  Who 393216
//   Qh  524288 (3,276,800)   outA 3,801,088 (3,276,800)
//   Kh  7,077,888 (25,165,824)  Vt 32,243,712 (25,165,824)  -> ~57.5 MB total
// ---------------------------------------------------------------------------
extern "C" void kernel_launch(void* const* d_in, const int* in_sizes, int n_in,
                              void* d_out, int out_size, void* d_ws, size_t ws_size,
                              hipStream_t stream) {
  const float* comp  = (const float*)d_in[0];
  const float* ref   = (const float*)d_in[1];
  const float* Wq    = (const float*)d_in[2];
  const float* Wk    = (const float*)d_in[3];
  const float* Wv    = (const float*)d_in[4];
  const float* Wo    = (const float*)d_in[5];
  const float* gamma = (const float*)d_in[6];
  const float* beta  = (const float*)d_in[7];
  float* out = (float*)d_out;

  char* ws = (char*)d_ws;
  h16* Whq  = (h16*)(ws + 0);
  h16* Whk  = (h16*)(ws + 131072);
  h16* Whv  = (h16*)(ws + 262144);
  h16* Who  = (h16*)(ws + 393216);
  h16* Qh   = (h16*)(ws + 524288);
  h16* outA = (h16*)(ws + 3801088);
  h16* Kh   = (h16*)(ws + 7077888);
  h16* Vt   = (h16*)(ws + 32243712);

  wcvt_kernel<<<(CH * CH) / 256, 256, 0, stream>>>(Wq, Wk, Wv, Wo, Whq, Whk, Whv, Who);
  qproj_kernel<<<BN / 16, 128, 0, stream>>>(comp, Whq, Qh);
  kvproj_kernel<<<BL / 64, 256, 0, stream>>>(ref, Whk, Whv, Kh, Vt);
  attn_kernel<<<BATCH * HEADS * 2, 128, 0, stream>>>(Qh, Kh, Vt, outA);
  oproj_ln_kernel<<<BN / 64, 128, 0, stream>>>(outA, Who, comp, gamma, beta, out);
}
